// VisualEmbeddingLayer1_56831007261327
// MI455X (gfx1250) — matrix operand order repaired and checked by reading, not compiled, not run on hardware
//
#include <hip/hip_runtime.h>
#include <hip/hip_bf16.h>

#define BB 256
#define NN 192
#define DD 512
#define EE 1024
#define HH 512
#define KK 58
#define MM (BB * KK) /* 14848 = 928*16 */

typedef __attribute__((ext_vector_type(16))) _Float16 v16h;
typedef __attribute__((ext_vector_type(8)))  _Float16 v8h;
typedef __attribute__((ext_vector_type(8)))  float    v8f;

// ---------------- WMMA helpers ----------------

__device__ __forceinline__ v8f wmma16(v16h a, v16h b, v8f c) {
  // D = A(16x32 f16) * B(32x16 f16) + C(16x16 f32)
  return __builtin_amdgcn_wmma_f32_16x16x32_f16(
      /*neg_a=*/false, a, /*neg_b=*/false, b,
      /*c_mod=*/(short)0, c, /*reuse_a=*/false, /*reuse_b=*/false);
}

// A fragment from an LDS tile laid out [16][DD] row-major.
// ISA 16-bit A 16x32 layout: lanes 0-15 row=lane hold K=k0+0..7 (h0..7) and
// K=k0+16..23 (h8..15); lanes 16-31 row=lane-16 hold K=k0+8..15 / k0+24..31.
__device__ __forceinline__ v16h frag_a_lds(const _Float16* As, int k0, int lane) {
  const int row = lane & 15;
  const int off = (lane & 16) ? 8 : 0;
  const _Float16* p = As + row * DD + k0 + off;
  v8h lo = *(const v8h*)p;          // 16B ds load
  v8h hi = *(const v8h*)(p + 16);   // 16B ds load
  return __builtin_shufflevector(lo, hi, 0, 1, 2, 3, 4, 5, 6, 7,
                                 8, 9, 10, 11, 12, 13, 14, 15);
}

// B fragment straight from global. W stored row-major (out=col, in=k), ldk=DD.
// ISA 16-bit B 32x16 layout: lanes 0-15 col=lane hold K=k0+0..7 and k0+16..23;
// lanes 16-31 col=lane-16 hold K=k0+8..15 / k0+24..31.
__device__ __forceinline__ v16h frag_b_glb(const _Float16* __restrict__ W,
                                           int colbase, int k0, int lane) {
  const int col  = lane & 15;
  const int koff = (lane & 16) ? 8 : 0;
  const _Float16* p = W + (size_t)(colbase + col) * DD + k0 + koff;
  v8h lo = *(const v8h*)p;
  v8h hi = *(const v8h*)(p + 16);
  return __builtin_shufflevector(lo, hi, 0, 1, 2, 3, 4, 5, 6, 7,
                                 8, 9, 10, 11, 12, 13, 14, 15);
}

__device__ __forceinline__ void atomicMaxF(float* addr, float v) {
  // ordered-int trick; out is pre-initialized to -inf
  if (v >= 0.f) atomicMax((int*)addr, __float_as_int(v));
  else          atomicMin((unsigned int*)addr, __float_as_uint(v));
}

// ---------------- utility kernels ----------------

__global__ void k_fill(float* p, int n, float v) {
  int i = blockIdx.x * blockDim.x + threadIdx.x;
  if (i < n) p[i] = v;
}

__global__ void k_cvt_f16(const float* __restrict__ in, _Float16* __restrict__ out, int n) {
  int i = blockIdx.x * blockDim.x + threadIdx.x;
  if (i < n) out[i] = (_Float16)in[i];
}

// ---------------- top-K selection (one wave per batch) ----------------

__global__ void k_topk(const float* __restrict__ attn, int* __restrict__ idx) {
  const int b = blockIdx.x;
  const int lane = threadIdx.x; // 32 threads
  __shared__ float vals[NN];
  const float* row = attn + (size_t)b * (NN + 1) * (NN + 1) + 1; // [b,0,1:]
  for (int j = lane; j < NN; j += 32) vals[j] = row[j];
  __syncthreads();
  for (int i = 0; i < KK; ++i) {
    float best = -__builtin_inff();
    int bi = NN;
    for (int j = lane; j < NN; j += 32) {
      float v = vals[j];
      if (v > best) { best = v; bi = j; }
    }
    for (int s = 16; s > 0; s >>= 1) {
      float ov = __shfl_xor(best, s, 32);
      int   oi = __shfl_xor(bi, s, 32);
      if (ov > best || (ov == best && oi < bi)) { best = ov; bi = oi; }
    }
    if (lane == 0) { idx[b * KK + i] = bi; vals[bi] = -__builtin_inff(); }
    __syncthreads();
  }
}

// ---------------- gather + L2-normalize -> fp16 (one wave per row) ----------------

__global__ void k_gather_norm(const float* __restrict__ emb, const int* __restrict__ idx,
                              _Float16* __restrict__ b16) {
  const int wave = threadIdx.x >> 5;
  const int lane = threadIdx.x & 31;
  const int m = blockIdx.x * 8 + wave; // MM/8 blocks, exact
  const int b = m / KK;
  const int patch = idx[m];
  const float* src = emb + ((size_t)b * NN + patch) * DD + lane * 16;
  float x[16];
#pragma unroll
  for (int i = 0; i < 4; ++i) {
    float4 t = ((const float4*)src)[i];
    x[4 * i + 0] = t.x; x[4 * i + 1] = t.y;
    x[4 * i + 2] = t.z; x[4 * i + 3] = t.w;
  }
  float ss = 0.f;
#pragma unroll
  for (int i = 0; i < 16; ++i) ss += x[i] * x[i];
#pragma unroll
  for (int s = 16; s > 0; s >>= 1) ss += __shfl_xor(ss, s, 32);
  const float inv = 1.f / (sqrtf(ss) + 1e-8f);
  v8h h0, h1;
#pragma unroll
  for (int i = 0; i < 8; ++i) {
    h0[i] = (_Float16)(x[i] * inv);
    h1[i] = (_Float16)(x[8 + i] * inv);
  }
  v8h* dst = (v8h*)(b16 + (size_t)m * DD + lane * 16);
  dst[0] = h0;
  dst[1] = h1;
}

// ---------------- GEMM1: h_pre = base @ W1^T + b1 ----------------

__global__ void k_gemm1(const _Float16* __restrict__ A16, const _Float16* __restrict__ W116,
                        const float* __restrict__ b1, float* __restrict__ hpre) {
  __shared__ __align__(16) _Float16 As[16 * DD]; // 16 KB
  const int tid = threadIdx.x;
  const int mtile = blockIdx.x * 16;
  {
    const uint4* g = (const uint4*)(A16 + (size_t)mtile * DD);
    uint4* s = (uint4*)As;
    for (int i = tid; i < 16 * DD / 8; i += 256) s[i] = g[i];
  }
  __syncthreads();
  const int wave = tid >> 5, lane = tid & 31;
  const int colbase = blockIdx.y * 128 + wave * 16;
  v8f c = {};
  for (int k0 = 0; k0 < DD; k0 += 32) {
    v16h a = frag_a_lds(As, k0, lane);
    v16h bf = frag_b_glb(W116, colbase, k0, lane);
    c = wmma16(a, bf, c);
  }
  const int colb = colbase + (lane & 15);
  const int rbase = (lane & 16) ? 8 : 0;
  const float bias = b1[colb];
#pragma unroll
  for (int r = 0; r < 8; ++r)
    hpre[(size_t)(mtile + rbase + r) * HH + colb] = c[r] + bias;
}

// ---------------- batch-norm statistics / finalize / apply ----------------

__global__ void k_bn_stats(const float* __restrict__ hpre, float* __restrict__ sums) {
  const int t = threadIdx.x; // 256
  const int r0 = blockIdx.x * 256;
  float s1a = 0.f, s2a = 0.f, s1b = 0.f, s2b = 0.f;
  for (int r = 0; r < 256; ++r) {
    const float* row = hpre + (size_t)(r0 + r) * HH;
    float va = row[t], vb = row[t + 256];
    s1a += va; s2a += va * va;
    s1b += vb; s2b += vb * vb;
  }
  atomicAdd(&sums[t], s1a);        atomicAdd(&sums[HH + t], s2a);
  atomicAdd(&sums[t + 256], s1b);  atomicAdd(&sums[HH + t + 256], s2b);
}

__global__ void k_bn_finalize(float* __restrict__ sums, const float* __restrict__ gamma,
                              const float* __restrict__ beta) {
  const int c = threadIdx.x; // 512
  const float invM = 1.f / (float)MM;
  float mean = sums[c] * invM;
  float var = sums[HH + c] * invM - mean * mean;
  float sc = rsqrtf(var + 1e-5f) * gamma[c];
  float sh = beta[c] - mean * sc;
  sums[c] = sc;
  sums[HH + c] = sh;
}

__global__ void k_bn_apply(const float* __restrict__ hpre, const float* __restrict__ sums,
                           _Float16* __restrict__ h16) {
  const int g = blockIdx.x * blockDim.x + threadIdx.x;
  const int e0 = g * 2;
  if (e0 >= MM * HH) return;
  const int c = e0 & (HH - 1);
  float v0 = fmaxf(0.f, hpre[e0] * sums[c] + sums[HH + c]);
  float v1 = fmaxf(0.f, hpre[e0 + 1] * sums[c + 1] + sums[HH + c + 1]);
  h16[e0] = (_Float16)v0;
  h16[e0 + 1] = (_Float16)v1;
}

// ---------------- GEMM2+GEMM_fc fused + max-over-K ----------------

__global__ void k_gemm2(const _Float16* __restrict__ H16, const _Float16* __restrict__ W216,
                        const _Float16* __restrict__ A16, const _Float16* __restrict__ Wfc16,
                        const float* __restrict__ b2, const float* __restrict__ bfc,
                        float* __restrict__ out) {
  __shared__ __align__(16) _Float16 Hs[16 * DD]; // 16 KB
  __shared__ __align__(16) _Float16 Bs[16 * DD]; // 16 KB
  const int tid = threadIdx.x;
  const int mtile = blockIdx.x * 16;
  {
    const uint4* gh = (const uint4*)(H16 + (size_t)mtile * DD);
    const uint4* gb = (const uint4*)(A16 + (size_t)mtile * DD);
    uint4* sh = (uint4*)Hs;
    uint4* sb = (uint4*)Bs;
    for (int i = tid; i < 16 * DD / 8; i += 256) { sh[i] = gh[i]; sb[i] = gb[i]; }
  }
  __syncthreads();
  const int wave = tid >> 5, lane = tid & 31;
  const int colbase = blockIdx.y * 128 + wave * 16;
  v8f c = {};
  for (int k0 = 0; k0 < DD; k0 += 32) {
    c = wmma16(frag_a_lds(Hs, k0, lane), frag_b_glb(W216, colbase, k0, lane), c);
    c = wmma16(frag_a_lds(Bs, k0, lane), frag_b_glb(Wfc16, colbase, k0, lane), c);
  }
  const int colb = colbase + (lane & 15);
  const int rbase = (lane & 16) ? 8 : 0;
  const float bias = b2[colb] + bfc[colb];
#pragma unroll
  for (int r = 0; r < 8; ++r) {
    const int m = mtile + rbase + r;
    const int bb = m / KK;
    atomicMaxF(out + (size_t)bb * EE + colb, c[r] + bias);
  }
}

// ---------------- launch ----------------

extern "C" void kernel_launch(void* const* d_in, const int* in_sizes, int n_in,
                              void* d_out, int out_size, void* d_ws, size_t ws_size,
                              hipStream_t stream) {
  (void)in_sizes; (void)n_in; (void)out_size; (void)ws_size;
  const float* emb   = (const float*)d_in[0];
  const float* attn  = (const float*)d_in[1];
  const float* Wfc   = (const float*)d_in[2];
  const float* bfc   = (const float*)d_in[3];
  const float* W1    = (const float*)d_in[4];
  const float* b1    = (const float*)d_in[5];
  const float* gamma = (const float*)d_in[6];
  const float* beta  = (const float*)d_in[7];
  const float* W2    = (const float*)d_in[8];
  const float* b2    = (const float*)d_in[9];
  float* out = (float*)d_out;

  char* ws = (char*)d_ws;
  size_t o = 0;
  int* idx         = (int*)(ws + o);      o += (size_t)MM * 4;        // 59392 (16B aligned)
  _Float16* base16 = (_Float16*)(ws + o); o += (size_t)MM * DD * 2;   // 15.2 MB
  _Float16* wfc16  = (_Float16*)(ws + o); o += (size_t)EE * DD * 2;   // 1 MB
  _Float16* w116   = (_Float16*)(ws + o); o += (size_t)HH * DD * 2;   // 0.5 MB
  _Float16* w216   = (_Float16*)(ws + o); o += (size_t)EE * HH * 2;   // 1 MB
  float* hpre      = (float*)(ws + o);    o += (size_t)MM * HH * 4;   // 30.4 MB
  _Float16* h16    = (_Float16*)(ws + o); o += (size_t)MM * HH * 2;   // 15.2 MB
  float* sums      = (float*)(ws + o);    o += 2 * HH * 4;            // ~61 MB total

  k_fill<<<(BB * EE + 255) / 256, 256, 0, stream>>>(out, BB * EE, -__builtin_inff());
  k_fill<<<(2 * HH + 255) / 256, 256, 0, stream>>>(sums, 2 * HH, 0.f);

  k_cvt_f16<<<(EE * DD + 255) / 256, 256, 0, stream>>>(Wfc, wfc16, EE * DD);
  k_cvt_f16<<<(HH * DD + 255) / 256, 256, 0, stream>>>(W1, w116, HH * DD);
  k_cvt_f16<<<(EE * HH + 255) / 256, 256, 0, stream>>>(W2, w216, EE * HH);

  k_topk<<<BB, 32, 0, stream>>>(attn, idx);
  k_gather_norm<<<MM / 8, 256, 0, stream>>>(emb, idx, base16);

  dim3 g1(MM / 16, HH / 128); // 928 x 4
  k_gemm1<<<g1, 256, 0, stream>>>(base16, w116, b1, hpre);

  k_bn_stats<<<MM / 256, 256, 0, stream>>>(hpre, sums);
  k_bn_finalize<<<1, HH, 0, stream>>>(sums, gamma, beta);
  k_bn_apply<<<(MM * HH / 2 + 255) / 256, 256, 0, stream>>>(hpre, sums, h16);

  dim3 g2(MM / 16, EE / 128); // 928 x 8
  k_gemm2<<<g2, 256, 0, stream>>>(h16, w216, base16, wfc16, b2, bfc, out);
}
